// Block_3642132267470
// MI455X (gfx1250) — matrix operand order repaired and checked by reading, not compiled, not run on hardware
//
#include <hip/hip_runtime.h>
#include <hip/hip_bf16.h>
#include <stdint.h>

// ---------------------------------------------------------------------------
// Transformer block for MI455X (gfx1250, wave32, WMMA).
// Matmuls: v_wmma_f32_16x16x32_f16 (f32 accum).
// GEMM staging: global_load_async_to_lds_b128 (ASYNCcnt) + double-buffered LDS.
// ---------------------------------------------------------------------------

#define USE_ASYNC 1

typedef __attribute__((ext_vector_type(16))) _Float16 v16h;
typedef __attribute__((ext_vector_type(8)))  _Float16 v8h;
typedef __attribute__((ext_vector_type(8)))  float    v8f;

#define DEVINL __device__ __forceinline__

static DEVINL v16h cat8(v8h lo, v8h hi) {
    return __builtin_shufflevector(lo, hi, 0,1,2,3,4,5,6,7,8,9,10,11,12,13,14,15);
}

static DEVINL v8f wmma_f16(v16h a, v16h b, v8f c) {
    return __builtin_amdgcn_wmma_f32_16x16x32_f16(
        false, a, false, b, (short)0, c, false, false);
}

static DEVINL void async_copy_b128(const void* gbase, unsigned byte_off, void* lds_ptr) {
#if USE_ASYNC
    asm volatile("global_load_async_to_lds_b128 %0, %1, %2"
                 :
                 : "v"((unsigned)(uintptr_t)lds_ptr),
                   "v"(byte_off),
                   "s"((unsigned long long)(uintptr_t)gbase)
                 : "memory");
#else
    *(v8h*)lds_ptr = *(const v8h*)((const char*)gbase + byte_off);
#endif
}

static DEVINL void async_wait() {
#if USE_ASYNC
    asm volatile("s_wait_asynccnt 0x0" ::: "memory");
#endif
}

// ---------------------------------------------------------------------------
// fp32 [K,N] -> fp16 transposed [N,K]  (weights; LDS-tiled 32x32)
// ---------------------------------------------------------------------------
__global__ __launch_bounds__(256) void cvtT_f32_f16(const float* __restrict__ w,
                                                    _Float16* __restrict__ wt,
                                                    int K, int N) {
    __shared__ _Float16 t[32][33];
    const int n0 = blockIdx.x * 32, k0 = blockIdx.y * 32;
    const int tx = threadIdx.x & 31, ty = threadIdx.x >> 5;  // 32 x 8
#pragma unroll
    for (int j = 0; j < 4; ++j) {
        int r = ty + j * 8;  // k within tile
        t[r][tx] = (_Float16)w[(size_t)(k0 + r) * N + n0 + tx];
    }
    __syncthreads();
#pragma unroll
    for (int j = 0; j < 4; ++j) {
        int r = ty + j * 8;  // n within tile
        wt[(size_t)(n0 + r) * K + k0 + tx] = t[tx][r];
    }
}

// ---------------------------------------------------------------------------
// LayerNorm: one block per row (C = 768 = 3 * 256), writes f16
// ---------------------------------------------------------------------------
__global__ __launch_bounds__(256) void ln_f16(const float* __restrict__ x,
                                              const float* __restrict__ g,
                                              const float* __restrict__ b,
                                              _Float16* __restrict__ out) {
    const int C = 768;
    const int row = blockIdx.x, tid = threadIdx.x;
    const float* xr = x + (size_t)row * C;
    float v[3], s = 0.f, q = 0.f;
#pragma unroll
    for (int j = 0; j < 3; ++j) {
        v[j] = xr[tid + j * 256];
        s += v[j]; q += v[j] * v[j];
    }
#pragma unroll
    for (int off = 16; off > 0; off >>= 1) {
        s += __shfl_xor(s, off, 32);
        q += __shfl_xor(q, off, 32);
    }
    __shared__ float sm[8], sq[8];
    int wid = tid >> 5, lane = tid & 31;
    if (lane == 0) { sm[wid] = s; sq[wid] = q; }
    __syncthreads();
    float ts = 0.f, tq = 0.f;
#pragma unroll
    for (int j = 0; j < 8; ++j) { ts += sm[j]; tq += sq[j]; }
    float mu = ts * (1.f / C);
    float rstd = rsqrtf(tq * (1.f / C) - mu * mu + 1e-5f);
    _Float16* orow = out + (size_t)row * C;
#pragma unroll
    for (int j = 0; j < 3; ++j) {
        int e = tid + j * 256;
        orow[e] = (_Float16)((v[j] - mu) * rstd * g[e] + b[e]);
    }
}

// ---------------------------------------------------------------------------
// WMMA GEMM: C[M,N] = A[M,K](f16) * BT[N,K](f16, pre-transposed), f32 accum.
// Block tile 128x128, 256 threads = 8 waves (2 M x 4 N), wave tile 64x32.
// K-step 64, double-buffered LDS, async global->LDS staging.
// EPI: 0 = store f16, 2 = store f32 + residual, 3 = exact GELU -> f16
// ---------------------------------------------------------------------------
template <int EPI>
__global__ __launch_bounds__(256) void gemm_wmma_k(const _Float16* __restrict__ A,
                                                   const _Float16* __restrict__ BT,
                                                   _Float16* __restrict__ o16,
                                                   float* __restrict__ o32,
                                                   const float* __restrict__ res,
                                                   int M, int N, int K) {
    constexpr int KT = 64;   // K per staged tile (2 WMMA k-steps)
    constexpr int KP = 72;   // 64 + 8 pad (f16)
    __shared__ alignas(16) _Float16 As[2][128 * KP];
    __shared__ alignas(16) _Float16 Bs[2][128 * KP];

    const int tid = threadIdx.x;
    const int lane = tid & 31, w = tid >> 5;
    const int lh = lane & 15, kh = lane >> 4;
    const int wm = w >> 2, wn = w & 3;
    const int mbase = blockIdx.y * 128, nbase = blockIdx.x * 128;

    v8f acc[4][2];
#pragma unroll
    for (int mt = 0; mt < 4; ++mt)
#pragma unroll
        for (int nt = 0; nt < 2; ++nt) acc[mt][nt] = (v8f)(0.f);

    auto stage = [&](int k0, int buf) {
#pragma unroll
        for (int i = 0; i < 4; ++i) {
            int c = tid + i * 256;           // 1024 chunks of 8 f16
            int r = c >> 3, c8 = (c & 7) * 8;
            unsigned aoff = (unsigned)(((size_t)(mbase + r) * K + k0 + c8) * 2);
            unsigned boff = (unsigned)(((size_t)(nbase + r) * K + k0 + c8) * 2);
            async_copy_b128(A,  aoff, &As[buf][r * KP + c8]);
            async_copy_b128(BT, boff, &Bs[buf][r * KP + c8]);
        }
    };

    const int nk = K / KT;
    stage(0, 0);
    async_wait();
    __syncthreads();

    for (int kt = 0; kt < nk; ++kt) {
        const int cur = kt & 1;
        if (kt + 1 < nk) stage((kt + 1) * KT, cur ^ 1);

        const _Float16* as = As[cur];
        const _Float16* bs = Bs[cur];
#pragma unroll
        for (int ks = 0; ks < 2; ++ks) {
            v16h af[4];
#pragma unroll
            for (int mt = 0; mt < 4; ++mt) {
                const _Float16* p = as + (wm * 64 + mt * 16 + lh) * KP + ks * 32 + kh * 8;
                af[mt] = cat8(*(const v8h*)p, *(const v8h*)(p + 16));
            }
            v16h bf[2];
#pragma unroll
            for (int nt = 0; nt < 2; ++nt) {
                const _Float16* p = bs + (wn * 32 + nt * 16 + lh) * KP + ks * 32 + kh * 8;
                bf[nt] = cat8(*(const v8h*)p, *(const v8h*)(p + 16));
            }
#pragma unroll
            for (int mt = 0; mt < 4; ++mt)
#pragma unroll
                for (int nt = 0; nt < 2; ++nt)
                    acc[mt][nt] = wmma_f16(af[mt], bf[nt], acc[mt][nt]);
        }

        if (kt + 1 < nk) {
            async_wait();
            __syncthreads();
        }
    }

    // epilogue: C layout -> lane holds col lh, rows kh*8 + vi
#pragma unroll
    for (int mt = 0; mt < 4; ++mt) {
#pragma unroll
        for (int nt = 0; nt < 2; ++nt) {
            int gcol = nbase + wn * 32 + nt * 16 + lh;
#pragma unroll
            for (int vi = 0; vi < 8; ++vi) {
                int grow = mbase + wm * 64 + mt * 16 + kh * 8 + vi;
                size_t idx = (size_t)grow * N + gcol;
                float val = acc[mt][nt][vi];
                if (EPI == 0) {
                    o16[idx] = (_Float16)val;
                } else if (EPI == 2) {
                    o32[idx] = val + res[idx];
                } else if (EPI == 3) {
                    float gl = 0.5f * val * (1.f + erff(val * 0.70710678118654752f));
                    o16[idx] = (_Float16)gl;
                }
            }
        }
    }
}

// ---------------------------------------------------------------------------
// Flash attention, causal. grid = (T/64, H, B), 128 threads = 4 waves,
// each wave owns 16 query rows. HD = 64, f16 WMMA, f32 online softmax.
// qkv layout: [B*T, 3*768] f16 (q | k | v sections per head).
// ---------------------------------------------------------------------------
__global__ __launch_bounds__(128) void flash_attn_k(const _Float16* __restrict__ qkv,
                                                    _Float16* __restrict__ y) {
    constexpr int HDP = 72;
    __shared__ alignas(16) _Float16 Qs[64 * HDP];      // [q][d]
    __shared__ alignas(16) _Float16 Ks[64 * HDP];      // [k][d] == K^T B-frag ready
    __shared__ alignas(16) _Float16 Vs[64 * HDP];      // [d][k] (transposed)
    __shared__ alignas(16) _Float16 Ps[4 * 16 * HDP];  // per-wave P staging

    const int tid = threadIdx.x, lane = tid & 31, w = tid >> 5;
    const int lh = lane & 15, kh = lane >> 4;
    const int qt = blockIdx.x, h = blockIdx.y, b = blockIdx.z;
    const int qbase = qt * 64;
    const size_t rowbase = (size_t)b * 2048;
    const int qoff = h * 64, koff = 768 + h * 64, voff = 1536 + h * 64;

#pragma unroll
    for (int i = 0; i < 4; ++i) {
        int c = tid + i * 128;
        int r = c >> 3, d8 = (c & 7) * 8;
        v8h vq = *(const v8h*)(qkv + (rowbase + qbase + r) * 2304 + qoff + d8);
#pragma unroll
        for (int j = 0; j < 8; ++j) vq[j] = vq[j] * (_Float16)0.125f;  // 1/sqrt(64)
        *(v8h*)(Qs + r * HDP + d8) = vq;
    }
    __syncthreads();

    v16h qa[2];
#pragma unroll
    for (int ks = 0; ks < 2; ++ks) {
        const _Float16* p = Qs + (w * 16 + lh) * HDP + ks * 32 + kh * 8;
        qa[ks] = cat8(*(const v8h*)p, *(const v8h*)(p + 16));
    }

    v8f o[4];
    float m[8], l[8];
#pragma unroll
    for (int nt = 0; nt < 4; ++nt) o[nt] = (v8f)(0.f);
#pragma unroll
    for (int vi = 0; vi < 8; ++vi) { m[vi] = -3.0e38f; l[vi] = 0.f; }

    _Float16* Pw = Ps + w * 16 * HDP;

    for (int t = 0; t <= qt; ++t) {
        __syncthreads();
        const int kb = t * 64;
#pragma unroll
        for (int i = 0; i < 4; ++i) {
            int c = tid + i * 128;
            int r = c >> 3, d8 = (c & 7) * 8;
            const _Float16* src = qkv + (rowbase + kb + r) * 2304;
            v8h vk = *(const v8h*)(src + koff + d8);
            *(v8h*)(Ks + r * HDP + d8) = vk;
            v8h vv = *(const v8h*)(src + voff + d8);
#pragma unroll
            for (int j = 0; j < 8; ++j) Vs[(d8 + j) * HDP + r] = vv[j];
        }
        __syncthreads();

        v8f s[4];
#pragma unroll
        for (int nt = 0; nt < 4; ++nt) s[nt] = (v8f)(0.f);
#pragma unroll
        for (int nt = 0; nt < 4; ++nt) {
#pragma unroll
            for (int ks = 0; ks < 2; ++ks) {
                const _Float16* p = Ks + (nt * 16 + lh) * HDP + ks * 32 + kh * 8;
                v16h bk = cat8(*(const v8h*)p, *(const v8h*)(p + 16));
                s[nt] = wmma_f16(qa[ks], bk, s[nt]);
            }
        }
        if (t == qt) {  // causal mask, diagonal tile only
#pragma unroll
            for (int nt = 0; nt < 4; ++nt) {
                int col = kb + nt * 16 + lh;
#pragma unroll
                for (int vi = 0; vi < 8; ++vi) {
                    int row = qbase + w * 16 + kh * 8 + vi;
                    if (col > row) s[nt][vi] = -1.0e30f;
                }
            }
        }
        float mn[8], al[8];
#pragma unroll
        for (int vi = 0; vi < 8; ++vi) {
            float v = fmaxf(fmaxf(s[0][vi], s[1][vi]), fmaxf(s[2][vi], s[3][vi]));
#pragma unroll
            for (int off = 1; off < 16; off <<= 1) v = fmaxf(v, __shfl_xor(v, off, 32));
            mn[vi] = fmaxf(m[vi], v);
            al[vi] = __expf(m[vi] - mn[vi]);
        }
#pragma unroll
        for (int vi = 0; vi < 8; ++vi) {
            float sum = 0.f;
#pragma unroll
            for (int nt = 0; nt < 4; ++nt) {
                float pv = __expf(s[nt][vi] - mn[vi]);
                s[nt][vi] = pv;
                sum += pv;
            }
#pragma unroll
            for (int off = 1; off < 16; off <<= 1) sum += __shfl_xor(sum, off, 32);
            l[vi] = l[vi] * al[vi] + sum;
            m[vi] = mn[vi];
        }
#pragma unroll
        for (int nt = 0; nt < 4; ++nt)
#pragma unroll
            for (int vi = 0; vi < 8; ++vi) o[nt][vi] *= al[vi];

#pragma unroll
        for (int nt = 0; nt < 4; ++nt)
#pragma unroll
            for (int vi = 0; vi < 8; ++vi)
                Pw[(kh * 8 + vi) * HDP + nt * 16 + lh] = (_Float16)s[nt][vi];
        __syncthreads();

        v16h pa[2];
#pragma unroll
        for (int ks = 0; ks < 2; ++ks) {
            const _Float16* p = Pw + lh * HDP + ks * 32 + kh * 8;
            pa[ks] = cat8(*(const v8h*)p, *(const v8h*)(p + 16));
        }
#pragma unroll
        for (int nt = 0; nt < 4; ++nt) {
#pragma unroll
            for (int ks = 0; ks < 2; ++ks) {
                const _Float16* p = Vs + (nt * 16 + lh) * HDP + ks * 32 + kh * 8;
                v16h bv = cat8(*(const v8h*)p, *(const v8h*)(p + 16));
                o[nt] = wmma_f16(pa[ks], bv, o[nt]);
            }
        }
    }

#pragma unroll
    for (int vi = 0; vi < 8; ++vi) {
        float inv = 1.f / l[vi];
        int row = qbase + w * 16 + kh * 8 + vi;
#pragma unroll
        for (int nt = 0; nt < 4; ++nt)
            y[(rowbase + row) * 768 + h * 64 + nt * 16 + lh] = (_Float16)(o[nt][vi] * inv);
    }
}

// ---------------------------------------------------------------------------
// Host orchestration
// ---------------------------------------------------------------------------
extern "C" void kernel_launch(void* const* d_in, const int* in_sizes, int n_in,
                              void* d_out, int out_size, void* d_ws, size_t ws_size,
                              hipStream_t stream) {
    const float* x     = (const float*)d_in[0];
    const float* ln1_g = (const float*)d_in[1];
    const float* ln1_b = (const float*)d_in[2];
    const float* W_qkv = (const float*)d_in[3];
    const float* W_ap  = (const float*)d_in[4];
    const float* ln2_g = (const float*)d_in[5];
    const float* ln2_b = (const float*)d_in[6];
    const float* W_fc  = (const float*)d_in[7];
    const float* W_mp  = (const float*)d_in[8];

    const int R = 4096;  // B*T token rows
    char* ws = (char*)d_ws;
    // workspace layout (bytes); all weight copies stored TRANSPOSED [N][K] f16
    _Float16* WqT  = (_Float16*)(ws + 0);                  // 2304 x  768
    _Float16* WapT = (_Float16*)(ws + 3538944);            //  768 x  768
    _Float16* WfcT = (_Float16*)(ws + 4718592);            // 3072 x  768
    _Float16* WmpT = (_Float16*)(ws + 9437184);            //  768 x 3072
    _Float16* ln1o  = (_Float16*)(ws + 14155776);          // 4096 x  768 f16
    _Float16* qkv16 = (_Float16*)(ws + 20447232);          // 4096 x 2304 f16
    _Float16* y16   = (_Float16*)(ws + 39321600);          // 4096 x  768 f16
    float*    x1    = (float*)   (ws + 45613056);          // 4096 x  768 f32
    _Float16* ln2o  = (_Float16*)(ws + 58195968);          // 4096 x  768 f16
    _Float16* h16   = (_Float16*)(ws + 64487424);          // 4096 x 3072 f16

    // weights -> f16, transposed to [N][K]
    cvtT_f32_f16<<<dim3(2304/32,  768/32), 256, 0, stream>>>(W_qkv, WqT,   768, 2304);
    cvtT_f32_f16<<<dim3( 768/32,  768/32), 256, 0, stream>>>(W_ap,  WapT,  768,  768);
    cvtT_f32_f16<<<dim3(3072/32,  768/32), 256, 0, stream>>>(W_fc,  WfcT,  768, 3072);
    cvtT_f32_f16<<<dim3( 768/32, 3072/32), 256, 0, stream>>>(W_mp,  WmpT, 3072,  768);

    // ln1 -> qkv -> attention -> proj(+x)
    ln_f16<<<R, 256, 0, stream>>>(x, ln1_g, ln1_b, ln1o);
    gemm_wmma_k<0><<<dim3(2304/128, R/128), 256, 0, stream>>>(
        ln1o, WqT, qkv16, nullptr, nullptr, R, 2304, 768);
    flash_attn_k<<<dim3(2048/64, 12, 2), 128, 0, stream>>>(qkv16, y16);
    gemm_wmma_k<2><<<dim3(768/128, R/128), 256, 0, stream>>>(
        y16, WapT, nullptr, x1, x, R, 768, 768);

    // ln2 -> fc (+GELU) -> proj(+x1) -> out
    ln_f16<<<R, 256, 0, stream>>>(x1, ln2_g, ln2_b, ln2o);
    gemm_wmma_k<3><<<dim3(3072/128, R/128), 256, 0, stream>>>(
        ln2o, WfcT, h16, nullptr, nullptr, R, 3072, 768);
    gemm_wmma_k<2><<<dim3(768/128, R/128), 256, 0, stream>>>(
        h16, WmpT, nullptr, (float*)d_out, x1, R, 768, 3072);
}
